// BiLSTM_CRF_7473243095222
// MI455X (gfx1250) — compile-verified, matrix operand order checked by reading, not run on hardware
//
#include <hip/hip_runtime.h>
#include <hip/hip_bf16.h>
#include <stdint.h>

// Problem dimensions (fixed by the reference)
#define T_  512
#define B_  64
#define V_  32000
#define E_  256
#define H_  512
#define NT_ 64

typedef _Float16 f16;
typedef __attribute__((ext_vector_type(16))) _Float16 v16h;
typedef __attribute__((ext_vector_type(8)))  _Float16 v8h;
typedef __attribute__((ext_vector_type(8)))  float    v8f;

// ---------------------------------------------------------------------------
// WMMA helpers.
// 16-bit A/B fragment layout (CDNA5 ISA 7.12.2): lanes 0-15 hold row (lane&15),
// K bytes [kc .. kc+7] then [kc+16 .. kc+23]; lanes 16-31 hold the same row but
// K [kc+8..kc+15] and [kc+24..kc+31].  Two 16-byte loads per fragment.
// ---------------------------------------------------------------------------
__device__ __forceinline__ v16h load_frag16(const f16* __restrict__ rowbase,
                                            int kc, int lane) {
  const f16* p = rowbase + kc + ((lane & 16) ? 8 : 0);
  v8h lo = *(const v8h*)(p);
  v8h hi = *(const v8h*)(p + 16);
  return __builtin_shufflevector(lo, hi, 0, 1, 2, 3, 4, 5, 6, 7,
                                         8, 9, 10, 11, 12, 13, 14, 15);
}

__device__ __forceinline__ v8f wmma_f16(v16h a, v16h b, v8f c) {
  // D = A(16x32 f16) * B(32x16 f16) + C(16x16 f32)
  return __builtin_amdgcn_wmma_f32_16x16x32_f16(
      /*neg_a=*/false, a, /*neg_b=*/false, b,
      /*c_mod=*/(short)0, c, /*reuse_a=*/false, /*reuse_b=*/false);
}

// ---------------------------------------------------------------------------
// Small utility kernels
// ---------------------------------------------------------------------------
__global__ void fill_f32(float* __restrict__ p, float v, size_t n) {
  size_t i = (size_t)blockIdx.x * 256 + threadIdx.x;
  if (i < n) p[i] = v;
}

__global__ void fill_f16(f16* __restrict__ p, float v, size_t n) {
  size_t i = (size_t)blockIdx.x * 256 + threadIdx.x;
  if (i < n) p[i] = (f16)v;
}

__global__ void cast_f32_f16(const float* __restrict__ in, f16* __restrict__ out,
                             size_t n) {
  size_t i = (size_t)blockIdx.x * 256 + threadIdx.x;
  if (i < n) out[i] = (f16)in[i];
}

// Embedding gather: x[t*B+b, e] = emb[src[t*B+b], e]  (f32 -> f16)
__global__ void embed_gather(const int* __restrict__ src,
                             const float* __restrict__ emb,
                             f16* __restrict__ x) {
  size_t i = (size_t)blockIdx.x * 256 + threadIdx.x;  // over T*B*E
  size_t row = i / E_;
  int    e   = (int)(i % E_);
  int    tok = src[row];
  x[i] = (f16)emb[(size_t)tok * E_ + e];
}

// Pack concatenated LSTM weights:  Wcat[g, 0:din] = Wih[g], Wcat[g, din:] = Whh[g]
// plus fused bias = bih + bhh.  f32 -> f16.
__global__ void pack_wcat(const float* __restrict__ Wih,
                          const float* __restrict__ Whh,
                          const float* __restrict__ bih,
                          const float* __restrict__ bhh,
                          f16* __restrict__ wcat, float* __restrict__ bias,
                          int din, int ktot) {
  size_t idx = (size_t)blockIdx.x * 256 + threadIdx.x;
  size_t total = (size_t)4 * H_ * ktot;
  if (idx < total) {
    int g = (int)(idx / ktot);
    int k = (int)(idx % ktot);
    float v = (k < din) ? Wih[(size_t)g * din + k]
                        : Whh[(size_t)g * H_ + (k - din)];
    wcat[idx] = (f16)v;
  }
  if (idx < (size_t)4 * H_) bias[idx] = bih[idx] + bhh[idx];
}

// ---------------------------------------------------------------------------
// LSTM recurrent step (one launch per time step; kernel boundary = global
// barrier for the h(t-1) -> h(t) dependency).
//
// grid  = (16 hidden-blocks, 2 directions), block = 256 threads (8 wave32).
// Each WG owns hidden units [hb*32, hb*32+32) for ALL four gate chunks so the
// LSTM cell can be fused after an LDS exchange.  Wave w: chunk = w>>1
// (i/f/g/o), 16-column sub-tile = (w&1).  Each wave accumulates a 64x16 gate
// tile (4 WMMA per 32-wide K-chunk) over K = din + H (fused input+recurrent
// projection against the packed [Wih|Whh] weights).
// ---------------------------------------------------------------------------
__global__ void __launch_bounds__(256)
lstm_step(const f16* __restrict__ xin, int din,
          const f16* __restrict__ hprev_base, f16* __restrict__ hout_base,
          float* __restrict__ c_base,
          f16* __restrict__ y, int ystride,
          const f16* __restrict__ wcat_base, const float* __restrict__ bias_base,
          int ktot, const int* __restrict__ lengths, int s) {
  const int dir = blockIdx.y;
  const int t   = dir ? (T_ - 1 - s) : s;

  const f16*  hprev = hprev_base + (size_t)dir * B_ * H_;
  f16*        hout  = hout_base  + (size_t)dir * B_ * H_;
  float*      cst   = c_base     + (size_t)dir * B_ * H_;
  const f16*  wcat  = wcat_base  + (size_t)dir * 4 * H_ * ktot;
  const float* bias = bias_base  + (size_t)dir * 4 * H_;
  const int   ycol0 = dir * H_;

  const int tid   = threadIdx.x;
  const int lane  = tid & 31;
  const int wave  = tid >> 5;
  const int chunk = wave >> 1;           // 0..3 : i, f, g, o
  const int ncl   = (wave & 1) * 16;     // 16-col sub-tile within the 32 hidden
  const int hb    = blockIdx.x;          // hidden block: units [hb*32, hb*32+32)
  const int gcol  = chunk * H_ + hb * 32 + ncl + (lane & 15);  // this lane's gate row

  v8f acc0 = {}, acc1 = {}, acc2 = {}, acc3 = {};
  const f16* brow = wcat + (size_t)gcol * ktot;
  const f16* xrow = xin + (size_t)t * B_ * din;
  const int  ar   = lane & 15;

  for (int kc = 0; kc < ktot; kc += 32) {
    __builtin_prefetch(brow + kc + 64, 0, 0);          // global_prefetch_b8
    v16h bf = load_frag16(brow, kc, lane);
    const f16* abase;
    int ald, kl;
    if (kc < din) { abase = xrow;  ald = din; kl = kc; }        // input part
    else          { abase = hprev; ald = H_;  kl = kc - din; }  // recurrent part
    v16h a0 = load_frag16(abase + (size_t)(0  + ar) * ald, kl, lane);
    v16h a1 = load_frag16(abase + (size_t)(16 + ar) * ald, kl, lane);
    v16h a2 = load_frag16(abase + (size_t)(32 + ar) * ald, kl, lane);
    v16h a3 = load_frag16(abase + (size_t)(48 + ar) * ald, kl, lane);
    acc0 = wmma_f16(a0, bf, acc0);
    acc1 = wmma_f16(a1, bf, acc1);
    acc2 = wmma_f16(a2, bf, acc2);
    acc3 = wmma_f16(a3, bf, acc3);
  }

  // Gate preactivations -> LDS (f32), grouped [chunk][batch][hidden-in-block].
  const float bv = bias[gcol];
  __shared__ float g_lds[4][B_][33];     // pad to 33 to dodge bank conflicts
  const int mofs = (lane & 16) ? 8 : 0;  // C layout: lanes 16-31 hold rows +8
  const int col  = ncl + (lane & 15);
#pragma unroll
  for (int r = 0; r < 8; ++r) {
    g_lds[chunk][ 0 + mofs + r][col] = acc0[r] + bv;
    g_lds[chunk][16 + mofs + r][col] = acc1[r] + bv;
    g_lds[chunk][32 + mofs + r][col] = acc2[r] + bv;
    g_lds[chunk][48 + mofs + r][col] = acc3[r] + bv;
  }
  __syncthreads();

  // Fused LSTM cell (packed-sequence semantics: freeze state / zero output
  // at padded steps).
  for (int idx = tid; idx < B_ * 32; idx += 256) {
    int b  = idx >> 5;
    int hh = idx & 31;
    int hg = hb * 32 + hh;
    float gi = g_lds[0][b][hh], gf = g_lds[1][b][hh];
    float gg = g_lds[2][b][hh], go = g_lds[3][b][hh];
    float si = 1.f / (1.f + __expf(-gi));
    float sf = 1.f / (1.f + __expf(-gf));
    float so = 1.f / (1.f + __expf(-go));
    float cold = cst[b * H_ + hg];
    float cnew = sf * cold + si * tanhf(gg);
    float hnew = so * tanhf(cnew);
    bool  msk  = t < lengths[b];
    float h2   = msk ? hnew : (float)hprev[b * H_ + hg];
    cst[b * H_ + hg]  = msk ? cnew : cold;
    hout[b * H_ + hg] = (f16)h2;
    y[((size_t)t * B_ + b) * ystride + ycol0 + hg] = (f16)(msk ? hnew : 0.f);
  }
}

// ---------------------------------------------------------------------------
// Generic WMMA GEMM:  out(MxN) = A(MxK f16, row-major) @ W^T (W is NxK f16,
// row-major) + bias.  WG tile = 64x128 (wave w owns a 64x16 slab).
// ---------------------------------------------------------------------------
template <bool OUT16>
__global__ void __launch_bounds__(256)
gemm_wmma(const f16* __restrict__ A, const f16* __restrict__ W,
          const float* __restrict__ bias, void* __restrict__ out,
          int M, int N, int K) {
  const int tid  = threadIdx.x;
  const int lane = tid & 31;
  const int wave = tid >> 5;
  const int n0   = blockIdx.x * 128 + wave * 16;
  if (n0 >= N) return;                       // wave-uniform; EXEC stays full
  const int m0 = blockIdx.y * 64;
  const int nc = n0 + (lane & 15);
  const int ar = lane & 15;

  const f16* brow = W + (size_t)nc * K;
  v8f acc0 = {}, acc1 = {}, acc2 = {}, acc3 = {};
  for (int kc = 0; kc < K; kc += 32) {
    __builtin_prefetch(brow + kc + 64, 0, 0);
    v16h bf = load_frag16(brow, kc, lane);
    v16h a0 = load_frag16(A + (size_t)(m0 +  0 + ar) * K, kc, lane);
    v16h a1 = load_frag16(A + (size_t)(m0 + 16 + ar) * K, kc, lane);
    v16h a2 = load_frag16(A + (size_t)(m0 + 32 + ar) * K, kc, lane);
    v16h a3 = load_frag16(A + (size_t)(m0 + 48 + ar) * K, kc, lane);
    acc0 = wmma_f16(a0, bf, acc0);
    acc1 = wmma_f16(a1, bf, acc1);
    acc2 = wmma_f16(a2, bf, acc2);
    acc3 = wmma_f16(a3, bf, acc3);
  }

  const float bv  = bias ? bias[nc] : 0.f;
  const int  mofs = (lane & 16) ? 8 : 0;
#pragma unroll
  for (int r = 0; r < 8; ++r) {
    float v0 = acc0[r] + bv, v1 = acc1[r] + bv;
    float v2 = acc2[r] + bv, v3 = acc3[r] + bv;
    size_t r0 = (size_t)(m0 +  0 + mofs + r) * N + nc;
    size_t r1 = (size_t)(m0 + 16 + mofs + r) * N + nc;
    size_t r2 = (size_t)(m0 + 32 + mofs + r) * N + nc;
    size_t r3 = (size_t)(m0 + 48 + mofs + r) * N + nc;
    if (OUT16) {
      f16* o = (f16*)out;
      o[r0] = (f16)v0; o[r1] = (f16)v1; o[r2] = (f16)v2; o[r3] = (f16)v3;
    } else {
      float* o = (float*)out;
      o[r0] = v0; o[r1] = v1; o[r2] = v2; o[r3] = v3;
    }
  }
}

// ---------------------------------------------------------------------------
// CRF negative log-likelihood (one WG of 64 threads per batch element;
// thread j owns tag j; trans staged in LDS).
// ---------------------------------------------------------------------------
__global__ void __launch_bounds__(64)
crf_forward(const float* __restrict__ feats, const int* __restrict__ tgt,
            const int* __restrict__ lens, const float* __restrict__ start,
            const float* __restrict__ endv, const float* __restrict__ trans,
            float* __restrict__ loss) {
  const int b = blockIdx.x, j = threadIdx.x;
  __shared__ float tr[NT_][NT_ + 1];
  __shared__ float ash[NT_ + 1];
  __shared__ float red[NT_];
  for (int i = j; i < NT_ * NT_; i += NT_) tr[i >> 6][i & 63] = trans[i];
  const int len = lens[b];
  float alpha = start[j] + feats[(size_t)b * NT_ + j];
  __syncthreads();

  for (int t = 1; t < T_; ++t) {
    ash[j] = alpha;
    __syncthreads();
    if (t < len) {                         // uniform per block
      float mx = -1e30f;
      for (int i = 0; i < NT_; ++i) mx = fmaxf(mx, ash[i] + tr[i][j]);
      float sm = 0.f;
      for (int i = 0; i < NT_; ++i) sm += __expf(ash[i] + tr[i][j] - mx);
      alpha = mx + __logf(sm) + feats[((size_t)t * B_ + b) * NT_ + j];
    }
    __syncthreads();
  }

  red[j] = alpha + endv[j];
  __syncthreads();
  if (j == 0) {
    float mx = -1e30f;
    for (int i = 0; i < NT_; ++i) mx = fmaxf(mx, red[i]);
    float sm = 0.f;
    for (int i = 0; i < NT_; ++i) sm += __expf(red[i] - mx);
    float denom = mx + __logf(sm);

    int   prev = tgt[b];
    float num  = start[prev] + feats[(size_t)b * NT_ + prev];
    for (int t = 1; t < T_; ++t) {
      int cur = tgt[(size_t)t * B_ + b];
      if (t < len)
        num += trans[prev * NT_ + cur] + feats[((size_t)t * B_ + b) * NT_ + cur];
      prev = cur;
    }
    num += endv[tgt[(size_t)(len - 1) * B_ + b]];
    atomicAdd(loss, denom - num);          // loss = -sum(num - denom)
  }
}

// ---------------------------------------------------------------------------
// Viterbi: forward max/argmax with int8 backpointers, then serial backtrace.
// ---------------------------------------------------------------------------
__global__ void __launch_bounds__(64)
viterbi_kernel(const float* __restrict__ feats, const int* __restrict__ lens,
               const float* __restrict__ start, const float* __restrict__ endv,
               const float* __restrict__ trans, signed char* __restrict__ bp,
               float* __restrict__ decoded) {
  const int b = blockIdx.x, j = threadIdx.x;
  __shared__ float tr[NT_][NT_ + 1];
  __shared__ float ssh[NT_ + 1];
  for (int i = j; i < NT_ * NT_; i += NT_) tr[i >> 6][i & 63] = trans[i];
  const int len = lens[b];
  float score = start[j] + feats[(size_t)b * NT_ + j];
  __syncthreads();

  for (int t = 1; t < T_; ++t) {
    ssh[j] = score;
    __syncthreads();
    float best = -1e30f;
    int   arg  = 0;
    for (int i = 0; i < NT_; ++i) {
      float v = ssh[i] + tr[i][j];
      if (v > best) { best = v; arg = i; }
    }
    bp[((size_t)t * B_ + b) * NT_ + j] = (signed char)arg;
    if (t < len) score = best + feats[((size_t)t * B_ + b) * NT_ + j];
    __syncthreads();
  }

  ssh[j] = score + endv[j];
  __syncthreads();
  if (j == 0) {
    int   cur  = 0;
    float best = ssh[0];
    for (int i = 1; i < NT_; ++i)
      if (ssh[i] > best) { best = ssh[i]; cur = i; }
    decoded[(size_t)(T_ - 1) * B_ + b] = (float)cur;
    for (int t = T_ - 1; t >= 1; --t) {
      if (t < len) cur = bp[((size_t)t * B_ + b) * NT_ + cur];
      decoded[(size_t)(t - 1) * B_ + b] = (float)cur;
    }
  }
}

// ---------------------------------------------------------------------------
// Host orchestration
// ---------------------------------------------------------------------------
extern "C" void kernel_launch(void* const* d_in, const int* in_sizes, int n_in,
                              void* d_out, int out_size, void* d_ws,
                              size_t ws_size, hipStream_t stream) {
  (void)in_sizes; (void)n_in; (void)out_size; (void)ws_size;

  const int*   src  = (const int*)d_in[0];
  const int*   tgt  = (const int*)d_in[1];
  const int*   lens = (const int*)d_in[2];
  const float* emb  = (const float*)d_in[3];
  const float* lp[16];
  for (int i = 0; i < 16; ++i) lp[i] = (const float*)d_in[4 + i];  // (Wih,Whh,bih,bhh) x 4
  const float* W0 = (const float*)d_in[20];
  const float* b0 = (const float*)d_in[21];
  const float* W1 = (const float*)d_in[22];
  const float* b1 = (const float*)d_in[23];
  const float* W2 = (const float*)d_in[24];
  const float* b2 = (const float*)d_in[25];
  const float* c_start = (const float*)d_in[26];
  const float* c_end   = (const float*)d_in[27];
  const float* c_trans = (const float*)d_in[28];

  // Workspace carve-out
  char*  w   = (char*)d_ws;
  size_t off = 0;
  auto alloc = [&](size_t bytes) {
    size_t o = off;
    off += (bytes + 255) & ~(size_t)255;
    return o;
  };
  const size_t MROWS = (size_t)T_ * B_;  // 32768
  const int ktot0 = E_ + H_;             // 768
  const int ktot1 = 2 * H_ + H_;         // 1536

  f16*   X1    = (f16*)(w + alloc(MROWS * 1024 * sizeof(f16)));
  f16*   X2    = (f16*)(w + alloc(MROWS * 1024 * sizeof(f16)));
  f16*   wcat0 = (f16*)(w + alloc((size_t)2 * 4 * H_ * ktot0 * sizeof(f16)));
  f16*   wcat1 = (f16*)(w + alloc((size_t)2 * 4 * H_ * ktot1 * sizeof(f16)));
  float* bias0 = (float*)(w + alloc((size_t)2 * 4 * H_ * sizeof(float)));
  float* bias1 = (float*)(w + alloc((size_t)2 * 4 * H_ * sizeof(float)));
  f16*   hbufA = (f16*)(w + alloc((size_t)2 * B_ * H_ * sizeof(f16)));
  f16*   hbufB = (f16*)(w + alloc((size_t)2 * B_ * H_ * sizeof(f16)));
  float* cbuf  = (float*)(w + alloc((size_t)2 * B_ * H_ * sizeof(float)));
  f16*   W0h   = (f16*)(w + alloc((size_t)H_ * 2 * H_ * sizeof(f16)));
  f16*   W1h   = (f16*)(w + alloc((size_t)(H_ / 2) * H_ * sizeof(f16)));
  f16*   W2h   = (f16*)(w + alloc((size_t)NT_ * (H_ / 2) * sizeof(f16)));
  f16*   A1    = (f16*)(w + alloc(MROWS * H_ * sizeof(f16)));
  f16*   A2    = (f16*)(w + alloc(MROWS * (H_ / 2) * sizeof(f16)));
  signed char* bp = (signed char*)(w + alloc(MROWS * NT_));

  float* feats   = (float*)d_out;
  float* loss    = feats + MROWS * NT_;
  float* decoded = loss + 1;

  // --- weight packing / casting -------------------------------------------
  {
    size_t n0 = (size_t)4 * H_ * ktot0;
    unsigned g0 = (unsigned)((n0 + 255) / 256);
    pack_wcat<<<g0, 256, 0, stream>>>(lp[0], lp[1], lp[2], lp[3],
                                      wcat0, bias0, E_, ktot0);
    pack_wcat<<<g0, 256, 0, stream>>>(lp[4], lp[5], lp[6], lp[7],
                                      wcat0 + (size_t)4 * H_ * ktot0,
                                      bias0 + 4 * H_, E_, ktot0);
    size_t n1 = (size_t)4 * H_ * ktot1;
    unsigned g1 = (unsigned)((n1 + 255) / 256);
    pack_wcat<<<g1, 256, 0, stream>>>(lp[8], lp[9], lp[10], lp[11],
                                      wcat1, bias1, 2 * H_, ktot1);
    pack_wcat<<<g1, 256, 0, stream>>>(lp[12], lp[13], lp[14], lp[15],
                                      wcat1 + (size_t)4 * H_ * ktot1,
                                      bias1 + 4 * H_, 2 * H_, ktot1);
  }
  {
    size_t n;
    n = (size_t)H_ * 2 * H_;
    cast_f32_f16<<<(unsigned)((n + 255) / 256), 256, 0, stream>>>(W0, W0h, n);
    n = (size_t)(H_ / 2) * H_;
    cast_f32_f16<<<(unsigned)((n + 255) / 256), 256, 0, stream>>>(W1, W1h, n);
    n = (size_t)NT_ * (H_ / 2);
    cast_f32_f16<<<(unsigned)((n + 255) / 256), 256, 0, stream>>>(W2, W2h, n);
  }

  // --- embedding gather + output init -------------------------------------
  {
    size_t n = MROWS * E_;
    embed_gather<<<(unsigned)(n / 256), 256, 0, stream>>>(src, emb, X1);
    fill_f32<<<1, 256, 0, stream>>>(loss, 0.f, 1);
  }

  // --- BiLSTM layer 0: X1 (din=256) -> X2 (2H) -----------------------------
  {
    size_t nh = (size_t)2 * B_ * H_;
    fill_f16<<<(unsigned)((nh + 255) / 256), 256, 0, stream>>>(hbufA, 0.f, nh);
    fill_f32<<<(unsigned)((nh + 255) / 256), 256, 0, stream>>>(cbuf, 0.f, nh);
    for (int s = 0; s < T_; ++s) {
      f16* hp = (s & 1) ? hbufB : hbufA;
      f16* ho = (s & 1) ? hbufA : hbufB;
      lstm_step<<<dim3(16, 2), 256, 0, stream>>>(X1, E_, hp, ho, cbuf, X2,
                                                 2 * H_, wcat0, bias0, ktot0,
                                                 lens, s);
    }
  }

  // --- BiLSTM layer 1: X2 (din=1024) -> X1 (2H) ----------------------------
  {
    size_t nh = (size_t)2 * B_ * H_;
    fill_f16<<<(unsigned)((nh + 255) / 256), 256, 0, stream>>>(hbufA, 0.f, nh);
    fill_f32<<<(unsigned)((nh + 255) / 256), 256, 0, stream>>>(cbuf, 0.f, nh);
    for (int s = 0; s < T_; ++s) {
      f16* hp = (s & 1) ? hbufB : hbufA;
      f16* ho = (s & 1) ? hbufA : hbufB;
      lstm_step<<<dim3(16, 2), 256, 0, stream>>>(X2, 2 * H_, hp, ho, cbuf, X1,
                                                 2 * H_, wcat1, bias1, ktot1,
                                                 lens, s);
    }
  }

  // --- linear head: 1024 -> 512 -> 256 -> 64 (feats, f32 in d_out) ---------
  gemm_wmma<true><<<dim3(4, (unsigned)(MROWS / 64)), 256, 0, stream>>>(
      X1, W0h, b0, A1, (int)MROWS, H_, 2 * H_);
  gemm_wmma<true><<<dim3(2, (unsigned)(MROWS / 64)), 256, 0, stream>>>(
      A1, W1h, b1, A2, (int)MROWS, H_ / 2, H_);
  gemm_wmma<false><<<dim3(1, (unsigned)(MROWS / 64)), 256, 0, stream>>>(
      A2, W2h, b2, feats, (int)MROWS, NT_, H_ / 2);

  // --- CRF loss + Viterbi decode -------------------------------------------
  crf_forward<<<B_, NT_, 0, stream>>>(feats, tgt, lens, c_start, c_end,
                                      c_trans, loss);
  viterbi_kernel<<<B_, NT_, 0, stream>>>(feats, lens, c_start, c_end, c_trans,
                                         bp, decoded);
}
